// ScaledDotProductAttention_62766652064286
// MI455X (gfx1250) — matrix operand order repaired and checked by reading, not compiled, not run on hardware
//
#include <hip/hip_runtime.h>
#include <math.h>

// ---------------------------------------------------------------------------
// MI455X / gfx1250 fused attention pipeline (wave32, WMMA f16 -> f32)
//   1) q_proj = f16(queries @ Wq^T)      [B,2048,512]
//   2) k_proj = f16(values  @ Wk^T)      [B,2048,512]
//   3) S      = (q_proj @ k_proj^T)/sqrt(512)  fp32 [B,2048,2048]
//   4) P      = softmax(S) rows, written f16 in-place over S rows
//   5) out    = P @ values               fp32 [B,2048,2048]
// Matmuls: v_wmma_f32_16x16x32_f16, double-buffered LDS.
// f16-sourced tiles staged with global_load_async_to_lds_b128 (ASYNCcnt);
// f32-sourced tiles staged via regs (convert f32->f16) with batched loads.
// ---------------------------------------------------------------------------

typedef __attribute__((ext_vector_type(16))) _Float16 v16h;
typedef __attribute__((ext_vector_type(8)))  _Float16 v8h;
typedef __attribute__((ext_vector_type(4)))  _Float16 v4h;
typedef __attribute__((ext_vector_type(8)))  float    v8f;

#define BATCH 16
#define SQ    2048
#define SKV   2048
#define QD    1024
#define VD    2048
#define KD    512

#define BM 128
#define BN 128
#define BK 32
#define LDSK 40   // padded K stride (halves): 80B rows -> 16B-aligned chunks

// ASYNCcnt-tracked 16B global->LDS copy. LDS address = low 32 bits of the
// flat shared-memory address.
__device__ __forceinline__ void async_copy16(void* lds, const void* g) {
  unsigned loff = (unsigned)(unsigned long long)lds;
  asm volatile("global_load_async_to_lds_b128 %0, %1, off"
               :: "v"(loff), "v"(g) : "memory");
}
__device__ __forceinline__ void wait_async0() {
  asm volatile("s_wait_asynccnt 0x0" ::: "memory");
}

// C[b,m,n] = scale * sum_k A[b,m,k] * B[k,n]
//   A: row-major [M x K] (f32: reg-stage+convert; f16: async copy)
//   B: !B_KxN -> source row-major [N x K] (B^T: weights / k_proj)
//       B_KxN -> source row-major [K x N] (values), transposed on stage
// 256 threads = 8 waves (4M x 2N); block tile 128x128x32, wave tile 32x64,
// 2x4 C fragments -> 8 WMMAs per K-step, double-buffered LDS.
template<typename AT, typename BT, bool B_KxN, bool OUT_HALF>
__global__ __launch_bounds__(256) void wmma_gemm(
    const AT* __restrict__ A,  size_t aBatch, int lda,
    const BT* __restrict__ Bm, size_t bBatch, int ldb,
    void* __restrict__ Cv,     size_t cBatch, int ldc,
    int K, float scale)
{
  constexpr bool A_ASYNC = (sizeof(AT) == 2);
  constexpr bool B_ASYNC = (sizeof(BT) == 2) && !B_KxN;

  __shared__ __align__(16) _Float16 As[2][BM][LDSK];
  __shared__ __align__(16) _Float16 Bs[2][BN][LDSK];

  const int tid  = threadIdx.x;
  const int lane = tid & 31;
  const int wid  = tid >> 5;
  const int wm   = (wid >> 1) * 32;
  const int wn   = (wid & 1)  * 64;
  const int hi16 = lane >> 4;
  const int lrow = lane & 15;

  const int m0 = blockIdx.y * BM;
  const int n0 = blockIdx.x * BN;

  const AT* Ab = A  + (size_t)blockIdx.z * aBatch;
  const BT* Bb = Bm + (size_t)blockIdx.z * bBatch;

  v8f acc[2][4];
  const v8f zacc = {};
  #pragma unroll
  for (int i = 0; i < 2; ++i)
    #pragma unroll
    for (int j = 0; j < 4; ++j) acc[i][j] = zacc;

  float4 aReg[4], bReg[4];

  // ---- staging helpers (tile = 128 rows x 32 K) ----
  auto loadA = [&](int k0) {                     // f32 path: global -> regs
    if constexpr (!A_ASYNC) {
      #pragma unroll
      for (int i = 0; i < 4; ++i) {
        int idx = tid + i * 256;                 // 0..1023
        int r = idx >> 3, kc = (idx & 7) << 2;
        aReg[i] = *reinterpret_cast<const float4*>(
            Ab + (size_t)(m0 + r) * lda + (k0 + kc));
      }
    }
  };
  auto storeA = [&](int buf) {                   // f32 path: cvt regs -> LDS
    if constexpr (!A_ASYNC) {
      #pragma unroll
      for (int i = 0; i < 4; ++i) {
        int idx = tid + i * 256;
        int r = idx >> 3, kc = (idx & 7) << 2;
        v4h h;
        h[0] = (_Float16)aReg[i].x; h[1] = (_Float16)aReg[i].y;
        h[2] = (_Float16)aReg[i].z; h[3] = (_Float16)aReg[i].w;
        *reinterpret_cast<v4h*>(&As[buf][r][kc]) = h;
      }
    }
  };
  auto asyncA = [&](int k0, int buf) {           // f16 path: async global->LDS
    if constexpr (A_ASYNC) {
      #pragma unroll
      for (int i = 0; i < 2; ++i) {
        int idx = tid + i * 256;                 // 0..511
        int r = idx >> 2, kc = (idx & 3) << 3;
        async_copy16(&As[buf][r][kc],
                     Ab + (size_t)(m0 + r) * lda + (k0 + kc));
      }
    }
  };
  auto loadB = [&](int k0) {
    if constexpr (!B_ASYNC) {
      #pragma unroll
      for (int i = 0; i < 4; ++i) {
        int idx = tid + i * 256;
        if constexpr (!B_KxN) {
          int r = idx >> 3, kc = (idx & 7) << 2;
          bReg[i] = *reinterpret_cast<const float4*>(
              Bb + (size_t)(n0 + r) * ldb + (k0 + kc));
        } else {
          int kr = idx >> 5, nc = (idx & 31) << 2;
          bReg[i] = *reinterpret_cast<const float4*>(
              Bb + (size_t)(k0 + kr) * ldb + (n0 + nc));
        }
      }
    }
  };
  auto storeB = [&](int buf) {
    if constexpr (!B_ASYNC) {
      #pragma unroll
      for (int i = 0; i < 4; ++i) {
        int idx = tid + i * 256;
        if constexpr (!B_KxN) {
          int r = idx >> 3, kc = (idx & 7) << 2;
          v4h h;
          h[0] = (_Float16)bReg[i].x; h[1] = (_Float16)bReg[i].y;
          h[2] = (_Float16)bReg[i].z; h[3] = (_Float16)bReg[i].w;
          *reinterpret_cast<v4h*>(&Bs[buf][r][kc]) = h;
        } else {                                  // transpose scatter
          int kr = idx >> 5, nc = (idx & 31) << 2;
          Bs[buf][nc + 0][kr] = (_Float16)bReg[i].x;
          Bs[buf][nc + 1][kr] = (_Float16)bReg[i].y;
          Bs[buf][nc + 2][kr] = (_Float16)bReg[i].z;
          Bs[buf][nc + 3][kr] = (_Float16)bReg[i].w;
        }
      }
    }
  };
  auto asyncB = [&](int k0, int buf) {
    if constexpr (B_ASYNC) {
      #pragma unroll
      for (int i = 0; i < 2; ++i) {
        int idx = tid + i * 256;
        int r = idx >> 2, kc = (idx & 3) << 3;
        async_copy16(&Bs[buf][r][kc],
                     Bb + (size_t)(n0 + r) * ldb + (k0 + kc));
      }
    }
  };
  auto compute = [&](int p) {
    v16h afrag[2], bfrag[4];
    #pragma unroll
    for (int mi = 0; mi < 2; ++mi) {
      // 16-bit A 16x32: lanes 0-15 hold K {0-7,16-23}, lanes 16-31 {8-15,24-31}
      const _Float16* ap = &As[p][wm + mi * 16 + lrow][hi16 * 8];
      v8h lo = *reinterpret_cast<const v8h*>(ap);
      v8h hi = *reinterpret_cast<const v8h*>(ap + 16);
      afrag[mi] = __builtin_shufflevector(lo, hi,
          0,1,2,3,4,5,6,7,8,9,10,11,12,13,14,15);
    }
    #pragma unroll
    for (int ni = 0; ni < 4; ++ni) {
      // B 32x16: lanes 0-15 hold K {0-7,16-23}, lanes 16-31 {8-15,24-31}
      const _Float16* bp = &Bs[p][wn + ni * 16 + lrow][hi16 * 8];
      v8h lo = *reinterpret_cast<const v8h*>(bp);
      v8h hi = *reinterpret_cast<const v8h*>(bp + 16);
      bfrag[ni] = __builtin_shufflevector(lo, hi,
          0,1,2,3,4,5,6,7,8,9,10,11,12,13,14,15);
    }
    #pragma unroll
    for (int mi = 0; mi < 2; ++mi)
      #pragma unroll
      for (int ni = 0; ni < 4; ++ni)
        acc[mi][ni] = __builtin_amdgcn_wmma_f32_16x16x32_f16(
            false, afrag[mi], false, bfrag[ni],
            (short)0, acc[mi][ni], false, false);
  };

  // ---- prologue: stage tile 0 into buffer 0 ----
  loadA(0); loadB(0);
  asyncA(0, 0); asyncB(0, 0);
  storeA(0); storeB(0);
  if constexpr (A_ASYNC || B_ASYNC) wait_async0();
  __syncthreads();

  int p = 0;
  for (int k0 = 0; k0 < K; k0 += BK) {
    const int kn = k0 + BK;
    const bool more = kn < K;       // uniform across block
    if (more) {
      loadA(kn); loadB(kn);          // global loads issued before WMMAs
      asyncA(kn, p ^ 1); asyncB(kn, p ^ 1);
      if (kn + BK < K)
        __builtin_prefetch(Ab + (size_t)(m0 + (tid & 127)) * lda + (kn + BK), 0, 1);
    }
    compute(p);                      // 8x v_wmma on buffer p
    if (more) { storeA(p ^ 1); storeB(p ^ 1); }
    if constexpr (A_ASYNC || B_ASYNC) { if (more) wait_async0(); }
    __syncthreads();
    p ^= 1;
  }

  // ---- epilogue: C f32 fragment: VGPR r -> (M = r + 8*hi16, N = lrow) ----
  if constexpr (OUT_HALF) {
    _Float16* C = reinterpret_cast<_Float16*>(Cv) + (size_t)blockIdx.z * cBatch;
    #pragma unroll
    for (int mi = 0; mi < 2; ++mi)
      #pragma unroll
      for (int ni = 0; ni < 4; ++ni)
        #pragma unroll
        for (int r = 0; r < 8; ++r) {
          size_t row = (size_t)(m0 + wm + mi * 16 + r + 8 * hi16);
          size_t col = (size_t)(n0 + wn + ni * 16 + lrow);
          C[row * (size_t)ldc + col] = (_Float16)(acc[mi][ni][r] * scale);
        }
  } else {
    float* C = reinterpret_cast<float*>(Cv) + (size_t)blockIdx.z * cBatch;
    #pragma unroll
    for (int mi = 0; mi < 2; ++mi)
      #pragma unroll
      for (int ni = 0; ni < 4; ++ni)
        #pragma unroll
        for (int r = 0; r < 8; ++r) {
          size_t row = (size_t)(m0 + wm + mi * 16 + r + 8 * hi16);
          size_t col = (size_t)(n0 + wn + ni * 16 + lrow);
          C[row * (size_t)ldc + col] = acc[mi][ni][r] * scale;
        }
  }
}

// One block per score row; reads 2048 f32, writes 2048 f16 in-place over the
// row's first half (all reads complete before writes via reduction barriers).
__global__ __launch_bounds__(256) void softmax_rows(float* __restrict__ S) {
  float* row = S + (size_t)blockIdx.x * SKV;
  __shared__ float red[256];
  const int tid = threadIdx.x;

  float v[8];
  float m = -3.402823466e38f;
  #pragma unroll
  for (int i = 0; i < 8; ++i) { v[i] = row[tid + i * 256]; m = fmaxf(m, v[i]); }

  red[tid] = m; __syncthreads();
  for (int s = 128; s > 0; s >>= 1) {
    if (tid < s) red[tid] = fmaxf(red[tid], red[tid + s]);
    __syncthreads();
  }
  m = red[0]; __syncthreads();

  float sum = 0.f;
  #pragma unroll
  for (int i = 0; i < 8; ++i) { v[i] = __expf(v[i] - m); sum += v[i]; }

  red[tid] = sum; __syncthreads();
  for (int s = 128; s > 0; s >>= 1) {
    if (tid < s) red[tid] += red[tid + s];
    __syncthreads();
  }
  const float inv = 1.0f / red[0];
  __syncthreads();   // all reads of this row complete -> safe to overwrite

  _Float16* prow = reinterpret_cast<_Float16*>(row);  // stride 2*SKV halves
  #pragma unroll
  for (int i = 0; i < 8; ++i) prow[tid + i * 256] = (_Float16)(v[i] * inv);
}

extern "C" void kernel_launch(void* const* d_in, const int* in_sizes, int n_in,
                              void* d_out, int out_size, void* d_ws, size_t ws_size,
                              hipStream_t stream) {
  const float* values  = (const float*)d_in[0];  // [16,2048,2048]
  const float* queries = (const float*)d_in[1];  // [16,2048,1024]
  const float* Wq      = (const float*)d_in[2];  // [512,1024]
  const float* Wk      = (const float*)d_in[3];  // [512,2048]
  float* out = (float*)d_out;                    // [16,2048,2048] f32

  // workspace (320 MiB): q_proj f16 | k_proj f16 | S fp32 (P f16 in-place)
  _Float16* qp = (_Float16*)d_ws;
  _Float16* kp = qp + (size_t)BATCH * SQ * KD;
  float*    S  = (float*)((char*)d_ws + ((size_t)64 << 20));

  const dim3 blk(256);

  // 1) q_proj = queries @ Wq^T
  wmma_gemm<float, float, false, true>
      <<<dim3(KD / BN, SQ / BM, BATCH), blk, 0, stream>>>(
      queries, (size_t)SQ * QD, QD,
      Wq,      (size_t)0,       QD,
      qp,      (size_t)SQ * KD, KD,
      QD, 1.0f);

  // 2) k_proj = values @ Wk^T
  wmma_gemm<float, float, false, true>
      <<<dim3(KD / BN, SKV / BM, BATCH), blk, 0, stream>>>(
      values, (size_t)SKV * VD, VD,
      Wk,     (size_t)0,        VD,
      kp,     (size_t)SKV * KD, KD,
      VD, 1.0f);

  // 3) S = (q_proj @ k_proj^T) / sqrt(512)   (async f16 staging)
  wmma_gemm<_Float16, _Float16, false, false>
      <<<dim3(SKV / BN, SQ / BM, BATCH), blk, 0, stream>>>(
      qp, (size_t)SQ * KD,   KD,
      kp, (size_t)SKV * KD,  KD,
      S,  (size_t)SQ * SKV,  SKV,
      KD, 0.044194173824159216f);

  // 4) P = softmax(S) rows, f16 in-place (row stride 2*SKV halves)
  softmax_rows<<<dim3(BATCH * SQ), blk, 0, stream>>>(S);

  // 5) out = P @ values  (A = f16 async; B = f32 transpose-stage)
  wmma_gemm<_Float16, float, true, false>
      <<<dim3(VD / BN, SQ / BM, BATCH), blk, 0, stream>>>(
      (const _Float16*)S, (size_t)SQ * SKV * 2, SKV * 2,
      values,             (size_t)SKV * VD,     VD,
      out,                (size_t)SQ * VD,      VD,
      SKV, 1.0f);
}
